// KernelSmoothedCombiner_45990509805730
// MI455X (gfx1250) — compile-verified, matrix operand order checked by reading, not run on hardware
//
#include <hip/hip_runtime.h>
#include <hip/hip_bf16.h>
#include <math.h>

// Problem constants (match the reference)
#define B_   4
#define S_   512
#define K_   64
#define D_   1024
#define V_   32000
#define NTOK (B_ * S_)      // 2048 tokens
#define TWOD (2 * D_)       // 2048

typedef float    v2f  __attribute__((ext_vector_type(2)));
typedef float    v8f  __attribute__((ext_vector_type(8)));
typedef _Float16 v16h __attribute__((ext_vector_type(16)));
typedef int      vi4  __attribute__((vector_size(4 * sizeof(int))));  // b128 payload

#define GAS __attribute__((address_space(1)))
#define LAS __attribute__((address_space(3)))

#if defined(__has_builtin)
#  if __has_builtin(__builtin_amdgcn_global_load_async_to_lds_b128)
#    define HAVE_ASYNC_LDS 1
#  endif
#  if __has_builtin(__builtin_amdgcn_wmma_f32_16x16x4_f32)
#    define HAVE_WMMA_F32 1
#  endif
#endif

// ---------------------------------------------------------------------------
// Kernel 0: zero-fill knn_probs (262 MB) with B128 stores, grid-stride.
// ---------------------------------------------------------------------------
__global__ void k_zero_probs(float4* __restrict__ out, long n4) {
  long i      = (long)blockIdx.x * blockDim.x + threadIdx.x;
  long stride = (long)gridDim.x * blockDim.x;
  float4 z = make_float4(0.f, 0.f, 0.f, 0.f);
  for (; i < n4; i += stride) out[i] = z;
}

// ---------------------------------------------------------------------------
// Kernel 1: one workgroup (256 thr = 8 wave32) per token.
//   Stage keys[tok] (64x1024 f32 = 256KB) into LDS via async global->LDS,
//   then: mean key -> bandwidth -> softmax(K) -> scatter-add into probs
//   -> weighted key sum to workspace. Keys read from HBM exactly once.
//   Each thread owns 4 CONTIGUOUS columns (d = 4t..4t+3) so every LDS row
//   access is a single ds_load_b128 and all global accesses are float4.
// ---------------------------------------------------------------------------
__global__ void __launch_bounds__(256, 1)
k_token(const float* __restrict__ q,     const float* __restrict__ keys,
        const int*   __restrict__ vals,  const float* __restrict__ dist,
        const float* __restrict__ bw_w,  const float* __restrict__ bw_b,
        float* __restrict__ probs,       float* __restrict__ wsk_out)
{
  extern __shared__ float keysLds[];          // K_ * D_ floats = 256 KB
  __shared__ float sarr[K_];
  __shared__ float wts[K_];
  __shared__ float red[8];
  __shared__ float s_bw;

  const int tok = blockIdx.x;
  const int t   = threadIdx.x;
  const float* krow = keys + (size_t)tok * (K_ * D_);

  // ---- stage keys -> LDS (16384 x b128 transfers, 64 per thread) ----
#ifdef HAVE_ASYNC_LDS
  #pragma unroll 4
  for (int i = 0; i < 64; ++i) {
    int e = t + 256 * i;                      // float4 index
    __builtin_amdgcn_global_load_async_to_lds_b128(
        (GAS vi4*)(krow + 4 * e), (LAS vi4*)(keysLds + 4 * e), 0, 0);
  }
#  if __has_builtin(__builtin_amdgcn_s_wait_asynccnt)
  __builtin_amdgcn_s_wait_asynccnt(0);
#  else
  asm volatile("s_wait_asynccnt 0" ::: "memory");
#  endif
#else
  #pragma unroll 4
  for (int i = 0; i < 64; ++i) {
    int e = t + 256 * i;
    ((float4*)keysLds)[e] = ((const float4*)krow)[e];
  }
#endif
  __syncthreads();

  const float4* kl4 = (const float4*)keysLds;        // [K_][256] float4
  const float4* q4  = (const float4*)(q + (size_t)tok * D_);

  // ---- mean over K for this thread's 4 columns (ds_load_b128 per row) ----
  float4 qv = q4[t];
  float4 s4 = make_float4(0.f, 0.f, 0.f, 0.f);
  for (int k = 0; k < K_; ++k) {
    float4 r = kl4[k * 256 + t];
    s4.x += r.x; s4.y += r.y; s4.z += r.z; s4.w += r.w;
  }
  float4 av = make_float4(s4.x * (1.0f / K_), s4.y * (1.0f / K_),
                          s4.z * (1.0f / K_), s4.w * (1.0f / K_));

  // ---- bandwidth dot: x = [q ; avg_key] against bw_w ----
  float4 bwq = ((const float4*)bw_w)[t];
  float4 bwa = ((const float4*)bw_w)[256 + t];
  float p = qv.x * bwq.x + qv.y * bwq.y + qv.z * bwq.z + qv.w * bwq.w
          + av.x * bwa.x + av.y * bwa.y + av.z * bwa.z + av.w * bwa.w;

  // wave32 reduce, then cross-wave via LDS
  #pragma unroll
  for (int off = 16; off > 0; off >>= 1) p += __shfl_xor(p, off, 32);
  if ((t & 31) == 0) red[t >> 5] = p;
  __syncthreads();
  if (t == 0) {
    float s = red[0];
    #pragma unroll
    for (int i = 1; i < 8; ++i) s += red[i];
    s_bw = expf(s + bw_b[0]);                 // bandwidth = exp(linear)
  }
  __syncthreads();
  const float bw = s_bw;

  // ---- softmax over K neighbors: -sqrt(d)/h ----
  if (t < K_) sarr[t] = -sqrtf(dist[(size_t)tok * K_ + t]) / bw;
  __syncthreads();
  float mx = -1e30f;
  for (int k = 0; k < K_; ++k) mx = fmaxf(mx, sarr[k]);   // LDS broadcast
  float se = 0.f;
  for (int k = 0; k < K_; ++k) se += expf(sarr[k] - mx);
  const float inv = 1.0f / se;
  if (t < K_) {
    float w = expf(sarr[t] - mx) * inv;
    wts[t] = w;
    // scatter-add into vocab row (duplicates possible -> f32 global atomic)
    atomicAdd(probs + (size_t)tok * V_ + vals[(size_t)tok * K_ + t], w);
  }
  __syncthreads();

  // ---- weighted key sum from LDS -> workspace (float4 in/out) ----
  float4 acc = make_float4(0.f, 0.f, 0.f, 0.f);
  for (int k = 0; k < K_; ++k) {
    float4 r = kl4[k * 256 + t];
    float w  = wts[k];
    acc.x = fmaf(w, r.x, acc.x); acc.y = fmaf(w, r.y, acc.y);
    acc.z = fmaf(w, r.z, acc.z); acc.w = fmaf(w, r.w, acc.w);
  }
  ((float4*)wsk_out)[(size_t)tok * 256 + t] = acc;
}

// ---------------------------------------------------------------------------
// Kernel 2: lambda = sigmoid(relu([q;wsk] @ W1^T + b1) @ W2^T + b2)
//   16 tokens per workgroup; 16x1024x2048 GEMM tile via V_WMMA_F32_16X16X4.
//   Software-pipelined: next 4 k-steps' A/B fragments are loaded while the
//   current 4 WMMAs execute, so waits batch instead of serializing each WMMA.
// ---------------------------------------------------------------------------
__global__ void __launch_bounds__(256, 1)
k_mlp(const float* __restrict__ q,  const float* __restrict__ wsk,
      const float* __restrict__ w1, const float* __restrict__ b1,
      const float* __restrict__ w2, const float* __restrict__ b2,
      float* __restrict__ lam_out)
{
  extern __shared__ float Xt[];               // 16 x 2048 floats = 128 KB
  __shared__ float lam_lds[16];

  const int tb   = blockIdx.x;                // 128 blocks
  const int tok0 = tb * 16;
  const int t    = threadIdx.x;
  if (t < 16) lam_lds[t] = 0.0f;

  // stage X = [q ; wsk] rows for 16 tokens (float4, coalesced)
  float4* Xt4 = (float4*)Xt;
  #pragma unroll 4
  for (int i = 0; i < 32; ++i) {
    int e4 = t + 256 * i;                     // 8192 float4 total
    int m  = e4 >> 9;                         // 512 float4 per row
    int c4 = e4 & 511;
    float4 v = (c4 < 256)
        ? ((const float4*)(q   + (size_t)(tok0 + m) * D_))[c4]
        : ((const float4*)(wsk + (size_t)(tok0 + m) * D_))[c4 - 256];
    Xt4[e4] = v;
  }
  __syncthreads();                            // also publishes lam_lds init

  const int lane = t & 31;
  const int wv   = t >> 5;                    // wave id 0..7
  const int nloc = lane & 15;                 // N within 16-col tile
  const int hi   = lane >> 4;                 // lane half -> K-pair / M-half
  const int mrow = lane & 15;                 // A-matrix row for this lane
  const float* arow = Xt + mrow * TWOD;

  float myrow[8];
  #pragma unroll
  for (int v = 0; v < 8; ++v) myrow[v] = 0.f;

  // each wave owns 8 of the 64 output 16x16 tiles (1024 cols)
  for (int tt = 0; tt < 8; ++tt) {
    const int    tile  = wv * 8 + tt;
    const int    nglob = tile * 16 + nloc;
    const float* w1row = w1 + (size_t)nglob * TWOD;   // W1[n][*]
    __builtin_prefetch(w1row, 0, 1);                  // global_prefetch_b8

    v8f acc = {0.f, 0.f, 0.f, 0.f, 0.f, 0.f, 0.f, 0.f};
#ifdef HAVE_WMMA_F32
    // A 16x4: lane holds X[mrow][k0..k0+1], k0 = kb + u*4 + hi*2 (b64 loads)
    v2f aC[4], bC[4];
    #pragma unroll
    for (int u = 0; u < 4; ++u) {
      const int k0 = u * 4 + hi * 2;
      aC[u] = *(const v2f*)(arow  + k0);
      bC[u] = *(const v2f*)(w1row + k0);
    }
    for (int kb = 0; kb + 16 < TWOD; kb += 16) {
      v2f aN[4], bN[4];
      #pragma unroll
      for (int u = 0; u < 4; ++u) {           // prefetch next block's frags
        const int k0 = kb + 16 + u * 4 + hi * 2;
        aN[u] = *(const v2f*)(arow  + k0);
        bN[u] = *(const v2f*)(w1row + k0);
      }
      #pragma unroll
      for (int u = 0; u < 4; ++u)             // 4 WMMAs on current block
        acc = __builtin_amdgcn_wmma_f32_16x16x4_f32(
                  false, aC[u], false, bC[u], (short)0, acc, false, false);
      #pragma unroll
      for (int u = 0; u < 4; ++u) { aC[u] = aN[u]; bC[u] = bN[u]; }
    }
    #pragma unroll
    for (int u = 0; u < 4; ++u)               // final (peeled) block
      acc = __builtin_amdgcn_wmma_f32_16x16x4_f32(
                false, aC[u], false, bC[u], (short)0, acc, false, false);
#else
    // fallback: codegen-verified f16 WMMA (convert on the fly)
    for (int kb = 0; kb < TWOD; kb += 32) {
      v16h af, bf;
      #pragma unroll
      for (int h = 0; h < 16; ++h) {
        int Kk = (h < 8) ? (hi * 8 + h) : (16 + hi * 8 + (h - 8));
        af[h] = (_Float16)arow[kb + Kk];
        bf[h] = (_Float16)w1row[kb + Kk];
      }
      acc = __builtin_amdgcn_wmma_f32_16x16x32_f16(
                false, af, false, bf, (short)0, acc, false, false);
    }
#endif
    // fused epilogue: +b1, ReLU, * w2 -> per-row partial
    const float bb1 = b1[nglob];
    const float ww2 = w2[nglob];
    #pragma unroll
    for (int v = 0; v < 8; ++v) {
      float h = acc[v] + bb1;                 // C VGPR v holds row M = v + 8*hi
      h = fmaxf(h, 0.0f);
      myrow[v] = fmaf(h, ww2, myrow[v]);
    }
  }

  // reduce partials across lanes/waves into 16 per-token scalars (ds_add_f32)
  #pragma unroll
  for (int v = 0; v < 8; ++v) atomicAdd(&lam_lds[v + 8 * hi], myrow[v]);
  __syncthreads();

  if (t < 16) {
    float z = lam_lds[t] + b2[0];
    lam_out[tok0 + t] = 1.0f / (1.0f + expf(-z));   // sigmoid
  }
}

// ---------------------------------------------------------------------------
extern "C" void kernel_launch(void* const* d_in, const int* in_sizes, int n_in,
                              void* d_out, int out_size, void* d_ws, size_t ws_size,
                              hipStream_t stream) {
  (void)in_sizes; (void)n_in; (void)out_size; (void)ws_size;
  const float* q    = (const float*)d_in[0];
  const float* keys = (const float*)d_in[1];
  const int*   vals = (const int*)  d_in[2];
  const float* dist = (const float*)d_in[3];
  const float* bw_w = (const float*)d_in[4];
  const float* bw_b = (const float*)d_in[5];
  const float* w1   = (const float*)d_in[6];
  const float* b1   = (const float*)d_in[7];
  const float* w2   = (const float*)d_in[8];
  const float* b2   = (const float*)d_in[9];

  float* probs = (float*)d_out;                          // (B,S,V)
  float* lam   = (float*)d_out + (size_t)NTOK * V_;      // (B,S,1)
  float* wsk   = (float*)d_ws;                           // (B,S,D) scratch, 8 MB

  const long n4 = ((long)NTOK * V_) / 4;
  k_zero_probs<<<dim3(4096), dim3(256), 0, stream>>>((float4*)probs, n4);

  k_token<<<dim3(NTOK), dim3(256), K_ * D_ * sizeof(float), stream>>>(
      q, keys, vals, dist, bw_w, bw_b, probs, wsk);

  k_mlp<<<dim3(NTOK / 16), dim3(256), 16 * TWOD * sizeof(float), stream>>>(
      q, wsk, w1, b1, w2, b2, lam);
}
